// CustomLSTM_8856222564913
// MI455X (gfx1250) — matrix operand order repaired and checked by reading, not compiled, not run on hardware
//
#include <hip/hip_runtime.h>
#include <hip/hip_bf16.h>
#include <math.h>

// ---------------------------------------------------------------------------
// HiPPO-LSTM persistent kernel for MI455X (gfx1250, wave32, WMMA, 320KB LDS).
// ---------------------------------------------------------------------------

#define B_    32      // batch
#define S_    2048    // sequence
#define D_    512     // input dim
#define H_    512     // hidden dim
#define N_    256     // hippo dim
#define G_    2048    // 4*H gates
#define KTOT  1280    // concat K: 512 (x) + 256 (hippo) + 512 (h)
#define NKT   40      // KTOT / 32
#define NWG   32
#define TPB   256
#define MAXSC 1000.0f

typedef __attribute__((ext_vector_type(16))) __bf16 v16bf;
typedef __attribute__((ext_vector_type(8)))  __bf16 v8bf;
typedef __attribute__((ext_vector_type(4)))  __bf16 v4bf;
typedef __attribute__((ext_vector_type(8)))  float  v8f;

// ---------------- workspace layout (bytes) ----------------
#define WFRAG_OFF 0u                         // 128 ct * 40 kt * 512 bf16 = 5,242,880 B
#define AFRAG_OFF 5242880u                   // 16 ct * 8 kt * 512 bf16  =   131,072 B
#define BCOMB_OFF 5373952u                   // 2048 f32
#define WSUM_OFF  5382144u                   // 512 f32
#define BSUM_OFF  5384192u                   // 1 f32 (padded to 256)
#define H_OFF     5384448u                   // 2 * 32*512 f32 = 131,072 B
#define HIP_OFF   5515520u                   // 2 * 32*256 f32 =  65,536 B
#define SYNC_OFF  5581056u                   // counters
#define WS_BYTES  5581312u

// ---------------- LDS layout (bytes) ----------------
#define L_WFRAG   0          // 4*40*512 bf16 = 163,840
#define L_AFRAG   163840     // 8*512 bf16    =   8,192
#define L_Z       172032     // 32*1280 bf16  =  81,920
#define L_GATES   253952     // 4*32*16 f32   =   8,192
#define L_C       262144     // 32*16 f32     =   2,048
#define L_AH      264192     // 32*16 f32     =   2,048
#define L_WSUM    266240     // 512 f32       =   2,048
#define L_FSUM    268288     // 32 f32
#define L_SH      268416     // 32 f32
#define SMEM_BYTES 268544

static __device__ __forceinline__ v16bf cat16(v8bf lo, v8bf hi) {
  return __builtin_shufflevector(lo, hi, 0,1,2,3,4,5,6,7,8,9,10,11,12,13,14,15);
}
// A-fragment: per lane, 16 contiguous bf16 (two 16B chunks at +0, +8).
static __device__ __forceinline__ v16bf load_afrag(const __bf16* p) {
  return cat16(*(const v8bf*)(p), *(const v8bf*)(p + 8));
}
// B-fragment: per lane, 16 contiguous bf16 (two 16B chunks at +0, +8).
static __device__ __forceinline__ v16bf load_bfrag(const __bf16* p) {
  return cat16(*(const v8bf*)(p), *(const v8bf*)(p + 8));
}

// Branch-free transcendentals: gfx1250 has a hardware v_tanh_f32.
static __device__ __forceinline__ float fast_tanh(float x) {
#if __has_builtin(__builtin_amdgcn_tanhf)
  return __builtin_amdgcn_tanhf(x);
#elif __has_builtin(__builtin_amdgcn_tanh_f32)
  return __builtin_amdgcn_tanh_f32(x);
#else
  float e = __expf(2.0f * x);
  return 1.0f - 2.0f / (e + 1.0f);
#endif
}
static __device__ __forceinline__ float sigm(float x) { return 1.0f / (1.0f + __expf(-x)); }
static __device__ __forceinline__ float clipf(float x) { return fminf(fmaxf(x, -MAXSC), MAXSC); }

// Device-wide generation barrier (32 resident WGs).
static __device__ __forceinline__ void grid_barrier(unsigned* cnt, unsigned* gen) {
  __threadfence();
  __syncthreads();
  if (threadIdx.x == 0) {
    unsigned g = __hip_atomic_load(gen, __ATOMIC_RELAXED, __HIP_MEMORY_SCOPE_AGENT);
    unsigned old = __hip_atomic_fetch_add(cnt, 1u, __ATOMIC_ACQ_REL, __HIP_MEMORY_SCOPE_AGENT);
    if (old == NWG - 1) {
      __hip_atomic_store(cnt, 0u, __ATOMIC_RELAXED, __HIP_MEMORY_SCOPE_AGENT);
      __hip_atomic_fetch_add(gen, 1u, __ATOMIC_RELEASE, __HIP_MEMORY_SCOPE_AGENT);
    } else {
      while (__hip_atomic_load(gen, __ATOMIC_ACQUIRE, __HIP_MEMORY_SCOPE_AGENT) == g) {
        __builtin_amdgcn_s_sleep(1);
      }
    }
  }
  __syncthreads();
}

// ---------------------------------------------------------------------------
// Prep kernels (run once per launch, stream-ordered before the main kernel).
// ---------------------------------------------------------------------------

// Concatenated gate weights [W_ih | W_hh] -> bf16 B-fragments, layout
// [ct(128)][kt(40)][lane(32)][j(16)], element = Wcat[n = ct*16 + lane%16,
// k = kt*32 + (lane/16)*16 + j].
__global__ void prep_wfrag_k(const float* __restrict__ Wih,
                             const float* __restrict__ Whh,
                             __bf16* __restrict__ wf) {
  for (size_t idx = (size_t)blockIdx.x * blockDim.x + threadIdx.x;
       idx < (size_t)128 * NKT * 512; idx += (size_t)gridDim.x * blockDim.x) {
    int j    = (int)(idx & 15);
    int lane = (int)((idx >> 4) & 31);
    size_t r = idx >> 9;
    int kt   = (int)(r % NKT);
    int ct   = (int)(r / NKT);
    int n = ct * 16 + (lane & 15);
    int k = kt * 32 + ((lane >> 4) * 16) + j;
    float v = (k < (D_ + N_)) ? Wih[(size_t)n * (D_ + N_) + k]
                              : Whh[(size_t)n * H_ + (k - (D_ + N_))];
    wf[idx] = (__bf16)v;
  }
}

// Fixed HiPPO A -> bf16 B-fragments (B[k,n] = A[n,k]),
// layout [ct(16)][kt(8)][lane(32)][j(16)].
__global__ void prep_afrag_k(const float* __restrict__ A, __bf16* __restrict__ af) {
  for (size_t idx = (size_t)blockIdx.x * blockDim.x + threadIdx.x;
       idx < (size_t)16 * 8 * 512; idx += (size_t)gridDim.x * blockDim.x) {
    int j    = (int)(idx & 15);
    int lane = (int)((idx >> 4) & 31);
    size_t r = idx >> 9;
    int kt   = (int)(r & 7);
    int ct   = (int)(r >> 3);
    int n = ct * 16 + (lane & 15);
    int k = kt * 32 + ((lane >> 4) * 16) + j;
    af[idx] = (__bf16)A[(size_t)n * N_ + k];
  }
}

// bcomb = b_ih + b_hh ; wsum[k] = sum_n W_h2h[n,k] ; bsum = sum_n b_h2h[n]
__global__ void prep_misc_k(const float* __restrict__ bih, const float* __restrict__ bhh,
                            const float* __restrict__ Wh2h, const float* __restrict__ bh2h,
                            float* __restrict__ bcomb, float* __restrict__ wsum,
                            float* __restrict__ bsum) {
  int tid = blockIdx.x * blockDim.x + threadIdx.x;
  if (tid < G_) bcomb[tid] = bih[tid] + bhh[tid];
  if (tid < H_) {
    float s = 0.0f;
    for (int n = 0; n < N_; ++n) s += Wh2h[(size_t)n * H_ + tid];
    wsum[tid] = s;
  }
  if (tid == 0) {
    float s = 0.0f;
    for (int n = 0; n < N_; ++n) s += bh2h[n];
    *bsum = s;
  }
}

__global__ void init_state_k(float* __restrict__ p, int n) {
  for (int i = blockIdx.x * blockDim.x + threadIdx.x; i < n; i += gridDim.x * blockDim.x)
    p[i] = 0.0f;
}

// ---------------------------------------------------------------------------
// Main persistent kernel: 32 WGs x 256 threads (8 wave32s each).
// WG wg owns hidden columns [16*wg, 16*wg+16) for all four gates.
// Wave wid: gate gt = wid>>1, batch-row tile mrow = wid&1.
// ---------------------------------------------------------------------------
__global__ __launch_bounds__(TPB, 1)
void hippo_lstm_main(const float* __restrict__ x,
                     float* __restrict__ out_h, float* __restrict__ out_c,
                     char* __restrict__ ws) {
  extern __shared__ char smem[];
  __bf16* wfrag = (__bf16*)(smem + L_WFRAG);
  __bf16* afrag = (__bf16*)(smem + L_AFRAG);
  __bf16* zl    = (__bf16*)(smem + L_Z);
  float*  gl    = (float*)(smem + L_GATES);
  float*  cl    = (float*)(smem + L_C);
  float*  ahl   = (float*)(smem + L_AH);
  float*  wsl   = (float*)(smem + L_WSUM);
  float*  fsl   = (float*)(smem + L_FSUM);
  float*  shl   = (float*)(smem + L_SH);

  const __bf16* gWfrag = (const __bf16*)(ws + WFRAG_OFF);
  const __bf16* gAfrag = (const __bf16*)(ws + AFRAG_OFF);
  const float*  gBcomb = (const float*)(ws + BCOMB_OFF);
  const float*  gWsum  = (const float*)(ws + WSUM_OFF);
  const float*  gBsum  = (const float*)(ws + BSUM_OFF);
  float*        hbuf   = (float*)(ws + H_OFF);
  float*        hipbuf = (float*)(ws + HIP_OFF);
  unsigned*     sync   = (unsigned*)(ws + SYNC_OFF);

  const int wg   = blockIdx.x;
  const int tid  = threadIdx.x;
  const int wid  = tid >> 5;
  const int lane = tid & 31;
  const int gt   = wid >> 1;   // gate 0..3 (i,f,g,o)
  const int mrow = wid & 1;    // batch rows [0,16) or [16,32)

  // ---- one-time staging: persist this WG's weight slice in LDS ----
  for (int g = 0; g < 4; ++g) {
    int ct = g * 32 + wg;                      // global column tile of gate g
    const int4* src = (const int4*)(gWfrag + (size_t)ct * NKT * 512);
    int4* dst = (int4*)(wfrag + (size_t)g * NKT * 512);
    for (int i = tid; i < NKT * 512 / 8; i += TPB) dst[i] = src[i];
  }
  if (wg < 16) {
    const int4* src = (const int4*)(gAfrag + (size_t)wg * 8 * 512);
    int4* dst = (int4*)afrag;
    for (int i = tid; i < 8 * 512 / 8; i += TPB) dst[i] = src[i];
  }
  for (int i = tid; i < H_; i += TPB) wsl[i] = gWsum[i];
  for (int i = tid; i < B_ * 16; i += TPB) cl[i] = 0.0f;
  const float bsum  = *gBsum;
  const float biasv = gBcomb[gt * H_ + wg * 16 + (lane & 15)];
  __syncthreads();

  const int arow = mrow * 16 + (lane & 15);   // batch row this lane feeds in A-frag
  const int ksh  = (lane >> 4) * 16;          // K-phase within 32-wide k-tile

  for (int t = 0; t < S_; ++t) {
    const int p = t & 1;
    const float* hippoR = hipbuf + p * (B_ * N_);
    float*       hippoW = hipbuf + (p ^ 1) * (B_ * N_);
    const float* hR     = hbuf + p * (B_ * H_);
    float*       hW     = hbuf + (p ^ 1) * (B_ * H_);

    // ---- stage z = [x_t | hippo | h] as bf16 in LDS ----
    for (int cc = tid; cc < B_ * (KTOT / 4); cc += TPB) {
      int b  = cc / (KTOT / 4);
      int k0 = (cc % (KTOT / 4)) * 4;
      float4 v;
      if (k0 < D_)            v = *(const float4*)(x + (size_t)b * S_ * D_ + (size_t)t * D_ + k0);
      else if (k0 < D_ + N_)  v = *(const float4*)(hippoR + b * N_ + (k0 - D_));
      else                    v = *(const float4*)(hR + b * H_ + (k0 - D_ - N_));
      v4bf o;
      o.x = (__bf16)v.x; o.y = (__bf16)v.y; o.z = (__bf16)v.z; o.w = (__bf16)v.w;
      *(v4bf*)(zl + (size_t)b * KTOT + k0) = o;
    }
    __syncthreads();

    // ---- gate GEMM tile: 40 x v_wmma_f32_16x16x32_bf16 per wave ----
    // Guard-free ping/pong pipeline, fully unrolled: the steady loop only
    // covers tiles [0, NKT-2) so prefetching kt+2/kt+3 is always in range;
    // the last two tiles are a branchless epilogue.  Full unroll folds all
    // LDS addresses into ds_load immediate offsets (max 40 KB < 64 KB field).
    v8f acc = {};
    {
      const __bf16* zrow = zl + (size_t)arow * KTOT;
      const __bf16* wp   = wfrag + (size_t)gt * NKT * 512 + lane * 16;
      v16bf a0 = load_afrag(zrow + ksh);
      v16bf b0 = load_bfrag(wp);
      v16bf a1 = load_afrag(zrow + 32 + ksh);
      v16bf b1 = load_bfrag(wp + 512);
#pragma unroll
      for (int kt = 0; kt < NKT - 2; kt += 2) {
        acc = __builtin_amdgcn_wmma_f32_16x16x32_bf16(false, a0, false, b0,
                                                      (short)0, acc, false, false);
        a0 = load_afrag(zrow + (kt + 2) * 32 + ksh);
        b0 = load_bfrag(wp + (kt + 2) * 512);
        acc = __builtin_amdgcn_wmma_f32_16x16x32_bf16(false, a1, false, b1,
                                                      (short)0, acc, false, false);
        a1 = load_afrag(zrow + (kt + 3) * 32 + ksh);
        b1 = load_bfrag(wp + (kt + 3) * 512);
      }
      acc = __builtin_amdgcn_wmma_f32_16x16x32_bf16(false, a0, false, b0,
                                                    (short)0, acc, false, false);
      acc = __builtin_amdgcn_wmma_f32_16x16x32_bf16(false, a1, false, b1,
                                                    (short)0, acc, false, false);
    }
    {
      int col = lane & 15;
      int r0  = mrow * 16 + ((lane >> 4) * 8);
      float* gp = gl + gt * (32 * 16);
#pragma unroll
      for (int r = 0; r < 8; ++r) gp[(r0 + r) * 16 + col] = acc[r] + biasv;
    }
    __syncthreads();

    // ---- LSTM pointwise: 32 rows x 16 cols, 2 elems/thread ----
#pragma unroll
    for (int u = 0; u < 2; ++u) {
      int e = tid * 2 + u;
      int b = e >> 4, col = e & 15;
      float ig = gl[0 * 512 + b * 16 + col];
      float fg = gl[1 * 512 + b * 16 + col];
      float gg = gl[2 * 512 + b * 16 + col];
      float og = gl[3 * 512 + b * 16 + col];
      float cn = sigm(fg) * cl[b * 16 + col] + sigm(ig) * fast_tanh(gg);
      float hn = sigm(og) * fast_tanh(cn);
      cl[b * 16 + col] = cn;
      int hcol = wg * 16 + col;
      hW[b * H_ + hcol] = hn;
      out_h[(size_t)b * S_ * H_ + (size_t)t * H_ + hcol] = hn;
      out_c[(size_t)b * S_ * H_ + (size_t)t * H_ + hcol] = cn;
    }
    // prefetch next timestep's x rows into cache (global_prefetch_b8)
    if (t + 1 < S_) {
      int b = tid >> 3, kseg = (tid & 7) * 64;
      __builtin_prefetch(x + (size_t)b * S_ * D_ + (size_t)(t + 1) * D_ + kseg, 0, 1);
    }

    grid_barrier(sync, sync + 1);   // h_new fully committed

    // ---- hippo update (WGs 0..15; wg owns hippo cols [16*wg,16*wg+16)) ----
    // hippo_new = clip( rowsum(hippo) - (hippo@A^T)/(t+1) + bvec * Fsum/(t+1) )
    if (wg < 16) {
      if (wid < 2) {  // 8 x WMMA for (hippo @ A^T) column tile, ping/pong
        v8f ac2 = {};
        const int row2 = wid * 16 + (lane & 15);
        const __bf16* zrow = zl + (size_t)row2 * KTOT + D_;   // bf16 hippo lives in z
        const __bf16* ap   = afrag + lane * 16;
        v16bf a0 = load_afrag(zrow + ksh);
        v16bf b0 = load_bfrag(ap);
        v16bf a1 = load_afrag(zrow + 32 + ksh);
        v16bf b1 = load_bfrag(ap + 512);
#pragma unroll
        for (int kt = 0; kt < 6; kt += 2) {
          ac2 = __builtin_amdgcn_wmma_f32_16x16x32_bf16(false, a0, false, b0,
                                                        (short)0, ac2, false, false);
          a0 = load_afrag(zrow + (kt + 2) * 32 + ksh);
          b0 = load_bfrag(ap + (kt + 2) * 512);
          ac2 = __builtin_amdgcn_wmma_f32_16x16x32_bf16(false, a1, false, b1,
                                                        (short)0, ac2, false, false);
          a1 = load_afrag(zrow + (kt + 3) * 32 + ksh);
          b1 = load_bfrag(ap + (kt + 3) * 512);
        }
        ac2 = __builtin_amdgcn_wmma_f32_16x16x32_bf16(false, a0, false, b0,
                                                      (short)0, ac2, false, false);
        ac2 = __builtin_amdgcn_wmma_f32_16x16x32_bf16(false, a1, false, b1,
                                                      (short)0, ac2, false, false);
        int col = lane & 15;
        int r0  = wid * 16 + ((lane >> 4) * 8);
#pragma unroll
        for (int r = 0; r < 8; ++r) ahl[(r0 + r) * 16 + col] = ac2[r];
      }
      if (tid < B_) {  // Fsum via rank-1 trick + hippo rowsum
        int b = tid;
        float fs = bsum;
        const float* hp = hW + b * H_;
        for (int k = 0; k < H_; ++k) fs += hp[k] * wsl[k];
        float sh = 0.0f;
        const float* pp = hippoR + b * N_;
        for (int k = 0; k < N_; ++k) sh += pp[k];
        fsl[b] = fs;
        shl[b] = sh;
      }
      __syncthreads();
      float invt = 1.0f / (float)(t + 1);
#pragma unroll
      for (int u = 0; u < 2; ++u) {
        int e = tid * 2 + u;
        int b = e >> 4, col = e & 15;
        int n = wg * 16 + col;
        float bv = 0.5f * sqrtf(2.0f * (float)n + 1.0f);
        float val = shl[b] - invt * ahl[b * 16 + col] + bv * invt * fsl[b];
        hippoW[b * N_ + n] = clipf(val);
      }
    }

    grid_barrier(sync, sync + 1);   // hippo_new committed before next staging
  }
}

// ---------------------------------------------------------------------------
extern "C" void kernel_launch(void* const* d_in, const int* in_sizes, int n_in,
                              void* d_out, int out_size, void* d_ws, size_t ws_size,
                              hipStream_t stream) {
  (void)in_sizes; (void)n_in; (void)out_size; (void)ws_size;
  const float* x    = (const float*)d_in[0];
  const float* Wih  = (const float*)d_in[1];
  const float* Whh  = (const float*)d_in[2];
  const float* bih  = (const float*)d_in[3];
  const float* bhh  = (const float*)d_in[4];
  const float* Wh2h = (const float*)d_in[5];
  const float* bh2h = (const float*)d_in[6];
  const float* A    = (const float*)d_in[7];
  // d_in[8] (B matrix) unused: columns are constant -> derived analytically.

  char* ws = (char*)d_ws;
  float* out_h = (float*)d_out;
  float* out_c = out_h + (size_t)B_ * S_ * H_;

  prep_wfrag_k<<<4096, 256, 0, stream>>>(Wih, Whh, (__bf16*)(ws + WFRAG_OFF));
  prep_afrag_k<<<256, 256, 0, stream>>>(A, (__bf16*)(ws + AFRAG_OFF));
  prep_misc_k<<<8, 256, 0, stream>>>(bih, bhh, Wh2h, bh2h,
                                     (float*)(ws + BCOMB_OFF),
                                     (float*)(ws + WSUM_OFF),
                                     (float*)(ws + BSUM_OFF));
  // zero h[2], hippo[2], sync counters
  init_state_k<<<64, 256, 0, stream>>>((float*)(ws + H_OFF),
                                       (int)((WS_BYTES - H_OFF) / 4));

  (void)hipFuncSetAttribute(reinterpret_cast<const void*>(hippo_lstm_main),
                            hipFuncAttributeMaxDynamicSharedMemorySize, SMEM_BYTES);
  hippo_lstm_main<<<NWG, TPB, SMEM_BYTES, stream>>>(x, out_h, out_c, ws);
}
